// Attention_21062519620320
// MI455X (gfx1250) — matrix operand order repaired and checked by reading, not compiled, not run on hardware
//
#include <hip/hip_runtime.h>
#include <hip/hip_bf16.h>

typedef __attribute__((ext_vector_type(16))) _Float16 v16h;
typedef __attribute__((ext_vector_type(8)))  _Float16 v8h;
typedef __attribute__((ext_vector_type(4)))  _Float16 v4h;
typedef __attribute__((ext_vector_type(8)))  float    v8f;

#define WMMA_F16(A, B, C) \
  __builtin_amdgcn_wmma_f32_16x16x32_f16(false, (A), false, (B), (short)0, (C), false, false)

// gfx1250 async global->LDS copy path (guarded; falls back to direct loads)
#if defined(__has_builtin)
#  if __has_builtin(__builtin_amdgcn_global_load_async_to_lds_b128)
#    define USE_ASYNC 1
#  endif
#endif

#ifdef USE_ASYNC
typedef __attribute__((ext_vector_type(4))) int v4i_async;
typedef __attribute__((address_space(1))) v4i_async* async_gp_t;  // global int4*
typedef __attribute__((address_space(3))) v4i_async* async_lp_t;  // LDS int4*
#  define ASYNC_CP16(SRC, DST) \
     __builtin_amdgcn_global_load_async_to_lds_b128( \
         (async_gp_t)(SRC), (async_lp_t)(DST), 0, 0)
#  if __has_builtin(__builtin_amdgcn_s_wait_asynccnt)
#    define WAIT_ASYNC(N) __builtin_amdgcn_s_wait_asynccnt(N)
#  else
#    define WAIT_ASYNC(N) asm volatile("s_wait_asynccnt %0" ::"i"(N) : "memory")
#  endif
#endif

// Problem constants
#define BATCH 2
#define SEQ   2048
#define DMODEL 1024
#define HEADS 16
#define DHEAD 64
#define ROWS  (BATCH * SEQ)         // 4096
#define SCALE 0.125f                // dh^-0.5 = 1/8

// ---------------------------------------------------------------------------
// Kernel 1: LayerNorm (f32 in) -> f16 out, one block per row of 1024
// ---------------------------------------------------------------------------
__global__ void ln_kernel(const float* __restrict__ x,
                          const float* __restrict__ gamma,
                          const float* __restrict__ beta,
                          _Float16* __restrict__ xn) {
  const int row = blockIdx.x;
  const int t = threadIdx.x;                 // 256 threads, 4 elems each
  const float* xr = x + (size_t)row * DMODEL;
  __shared__ float rs[256];
  __shared__ float rq[256];
  float4 v = ((const float4*)xr)[t];
  rs[t] = v.x + v.y + v.z + v.w;
  rq[t] = v.x * v.x + v.y * v.y + v.z * v.z + v.w * v.w;
  __syncthreads();
  for (int off = 128; off > 0; off >>= 1) {
    if (t < off) { rs[t] += rs[t + off]; rq[t] += rq[t + off]; }
    __syncthreads();
  }
  const float mean = rs[0] * (1.0f / DMODEL);
  const float var  = rq[0] * (1.0f / DMODEL) - mean * mean;
  const float rstd = rsqrtf(var + 1e-5f);
  const int i = t * 4;
  float4 g = ((const float4*)gamma)[t];
  float4 bb = ((const float4*)beta)[t];
  v4h o;
  o[0] = (_Float16)((v.x - mean) * rstd * g.x + bb.x);
  o[1] = (_Float16)((v.y - mean) * rstd * g.y + bb.y);
  o[2] = (_Float16)((v.z - mean) * rstd * g.z + bb.z);
  o[3] = (_Float16)((v.w - mean) * rstd * g.w + bb.w);
  *(v4h*)(xn + (size_t)row * DMODEL + i) = o;
}

// ---------------------------------------------------------------------------
// Kernel 2: transpose-convert weight f32 [K][Ncols] -> f16 [Ncols][K]
// ---------------------------------------------------------------------------
__global__ void convT_kernel(const float* __restrict__ src,
                             _Float16* __restrict__ dst,
                             int K, int Ncols) {
  int idx = blockIdx.x * 256 + threadIdx.x;
  if (idx >= K * Ncols) return;
  int c = idx / K;
  int k = idx - c * K;
  dst[idx] = (_Float16)src[(size_t)k * Ncols + c];
}

// ---------------------------------------------------------------------------
// Kernel 3: QKV GEMM  [4096 x 1024] x [1024 x 3072] via f16 WMMA.
// Wave tile = 16 x 64 (4 accumulators). B = wT f16 [3072][1024] (contiguous K).
// Outputs: q,k head-major [b,h,n,dh]; v transposed [b,h,dh,n].
// ---------------------------------------------------------------------------
__global__ void qkv_kernel(const _Float16* __restrict__ xn,
                           const _Float16* __restrict__ wT,
                           _Float16* __restrict__ qh,
                           _Float16* __restrict__ kh,
                           _Float16* __restrict__ vt) {
  const int NT = 3072 / 64;                       // 48 N-tiles
  const int w = blockIdx.x * 8 + (threadIdx.x >> 5);
  const int lane = threadIdx.x & 31;
  const int tm = w / NT, tn = w - tm * NT;
  const int hl = lane >> 4;                       // half-wave id
  const int l15 = lane & 15;
  const _Float16* Arow = xn + (size_t)(tm * 16 + l15) * DMODEL;
  const int kbA = hl * 16;
  const int kbB = hl * 16;
  v8f acc[4] = {};
  for (int k0 = 0; k0 < DMODEL; k0 += 32) {
    v16h a;
    v8h alo = *(const v8h*)(Arow + k0 + kbA);
    v8h ahi = *(const v8h*)(Arow + k0 + 8 + kbA);
#pragma unroll
    for (int i = 0; i < 8; i++) { a[i] = alo[i]; a[8 + i] = ahi[i]; }
#pragma unroll
    for (int c = 0; c < 4; c++) {
      const int col = tn * 64 + c * 16 + l15;
      v16h b = *(const v16h*)(wT + (size_t)col * DMODEL + k0 + kbB);
      acc[c] = WMMA_F16(a, b, acc[c]);
    }
  }
#pragma unroll
  for (int c = 0; c < 4; c++) {
    const int col = tn * 64 + c * 16 + l15;
    const int which = col >> 10;                  // 0=q 1=k 2=v
    const int dg = col & 1023;
    const int h = dg >> 6, d = dg & 63;
#pragma unroll
    for (int j = 0; j < 8; j++) {
      const int r = tm * 16 + j + hl * 8;
      const int b = r >> 11, n = r & (SEQ - 1);
      const _Float16 val = (_Float16)acc[c][j];
      const size_t bh = (size_t)(b * HEADS + h);
      if (which == 0)      qh[(bh * SEQ + n) * DHEAD + d] = val;
      else if (which == 1) kh[(bh * SEQ + n) * DHEAD + d] = val;
      else                 vt[(bh * DHEAD + d) * SEQ + n] = val;
    }
  }
}

// ---------------------------------------------------------------------------
// Kernel 4: flash attention. One wave per (b,h,16-query tile), 2 waves/block.
// scores = Q Kt (4 WMMAs / 32 keys), online softmax, P V (4 WMMAs / 32 keys).
// K/V tiles double-buffered in LDS via GLOBAL_LOAD_ASYNC_TO_LDS_B128, synced
// with s_wait_asynccnt (loads retire in order: wait<=16 == current buf done).
// P transits LDS to convert C-layout -> A-layout.
// ---------------------------------------------------------------------------
__global__ void attn_kernel(const _Float16* __restrict__ qh,
                            const _Float16* __restrict__ kh,
                            const _Float16* __restrict__ vt,
                            const float* __restrict__ alibi,
                            _Float16* __restrict__ oh) {
  __shared__ __align__(16) _Float16 sp[2][16 * 32];        // P tiles
#ifdef USE_ASYNC
  __shared__ __align__(16) _Float16 skb[2][2][32 * DHEAD]; // K: [key][d]
  __shared__ __align__(16) _Float16 svb[2][2][DHEAD * 32]; // V: [d][key]
#endif
  const int wi = threadIdx.x >> 5;
  const int lane = threadIdx.x & 31;
  const int gw = blockIdx.x * 2 + wi;             // 0..4095
  const int qt = gw & 127;
  const int bh = gw >> 7;                         // b*16 + h
  const int h = bh & (HEADS - 1);
  const int hl = lane >> 4, l15 = lane & 15;
  const int q0 = qt * 16;

  const _Float16* Q = qh + ((size_t)bh * SEQ + q0) * DHEAD;
  const _Float16* K = kh + (size_t)bh * SEQ * DHEAD;
  const _Float16* V = vt + (size_t)bh * DHEAD * SEQ;
  const float* AB = alibi + ((size_t)h * SEQ + q0) * SEQ;

  // Q A-fragments for K-dim chunks [0,32) and [32,64)
  v16h qa0, qa1;
  {
    const _Float16* qr = Q + (size_t)l15 * DHEAD;
    v8h t0 = *(const v8h*)(qr + hl * 16);
    v8h t1 = *(const v8h*)(qr + 8 + hl * 16);
    v8h t2 = *(const v8h*)(qr + 32 + hl * 16);
    v8h t3 = *(const v8h*)(qr + 40 + hl * 16);
#pragma unroll
    for (int i = 0; i < 8; i++) {
      qa0[i] = t0[i]; qa0[8 + i] = t1[i];
      qa1[i] = t2[i]; qa1[8 + i] = t3[i];
    }
  }

  float mrow[8], lrow[8];
  v8f o[4] = {};
#pragma unroll
  for (int j = 0; j < 8; j++) { mrow[j] = -1e30f; lrow[j] = 0.0f; }
  _Float16* P = &sp[wi][0];

#ifdef USE_ASYNC
  // per-lane copy coordinates: K tile 32x64 halves, V tile 64x32 halves
  const int kkey = lane >> 3;            // 0..3
  const int kch  = (lane & 7) * 8;       // halves within a 64-half K row
  const int vrow = lane >> 2;            // 0..7
  const int vch  = (lane & 3) * 8;       // halves within a 32-half V row
  {                                      // prologue: prefetch tile 0 -> buf 0
    _Float16* kd = &skb[wi][0][0];
    _Float16* vd = &svb[wi][0][0];
#pragma unroll
    for (int t = 0; t < 8; t++) {
      ASYNC_CP16(K + (size_t)(t * 4 + kkey) * DHEAD + kch,
                 kd + (t * 4 + kkey) * DHEAD + kch);
      ASYNC_CP16(V + (size_t)(t * 8 + vrow) * SEQ + vch,
                 vd + (t * 8 + vrow) * 32 + vch);
    }
  }
#endif

  for (int it = 0; it < SEQ / 32; ++it) {
    const int n0 = it * 32;
#ifdef USE_ASYNC
    const int cur = it & 1;
    // buffer cur^1 was fully consumed (dscnt-waited before WMMAs) last iter
    asm volatile("s_wait_dscnt 0" ::: "memory");
    if (n0 + 32 < SEQ) {
      _Float16* kd = &skb[wi][cur ^ 1][0];
      _Float16* vd = &svb[wi][cur ^ 1][0];
      const int m0 = n0 + 32;
#pragma unroll
      for (int t = 0; t < 8; t++) {
        ASYNC_CP16(K + (size_t)(m0 + t * 4 + kkey) * DHEAD + kch,
                   kd + (t * 4 + kkey) * DHEAD + kch);
        ASYNC_CP16(V + (size_t)(t * 8 + vrow) * SEQ + m0 + vch,
                   vd + (t * 8 + vrow) * 32 + vch);
      }
      WAIT_ASYNC(16);                    // oldest 16 (current buffer) done
    } else {
      WAIT_ASYNC(0);
    }
    __builtin_amdgcn_wave_barrier();
    const _Float16* Kt = &skb[wi][cur][0];
    const _Float16* Vt = &svb[wi][cur][0];
#endif
    // ---- scores: 16 queries x 32 keys (two 16-col sub-tiles) ----
    v8f s0 = {}, s1 = {};
    {
#ifdef USE_ASYNC
      const _Float16* kr0 = Kt + l15 * DHEAD;
      const _Float16* kr1 = Kt + (16 + l15) * DHEAD;
#else
      const _Float16* kr0 = K + (size_t)(n0 + l15) * DHEAD;
      const _Float16* kr1 = K + (size_t)(n0 + 16 + l15) * DHEAD;
#endif
      v16h kb;
      kb = *(const v16h*)(kr0 + hl * 16);      s0 = WMMA_F16(qa0, kb, s0);
      kb = *(const v16h*)(kr0 + 32 + hl * 16); s0 = WMMA_F16(qa1, kb, s0);
      kb = *(const v16h*)(kr1 + hl * 16);      s1 = WMMA_F16(qa0, kb, s1);
      kb = *(const v16h*)(kr1 + 32 + hl * 16); s1 = WMMA_F16(qa1, kb, s1);
    }
    // ---- scale + alibi (read exactly once from HBM) ----
    float sc0[8], sc1[8];
#pragma unroll
    for (int j = 0; j < 8; j++) {
      const float* ar = AB + (size_t)(j + hl * 8) * SEQ + n0;
      sc0[j] = s0[j] * SCALE + ar[l15];
      sc1[j] = s1[j] * SCALE + ar[16 + l15];
    }
    // ---- online softmax (row reduction across 16-lane half-wave) ----
#pragma unroll
    for (int j = 0; j < 8; j++) {
      float t = fmaxf(sc0[j], sc1[j]);
      t = fmaxf(t, __shfl_xor(t, 1, 32));
      t = fmaxf(t, __shfl_xor(t, 2, 32));
      t = fmaxf(t, __shfl_xor(t, 4, 32));
      t = fmaxf(t, __shfl_xor(t, 8, 32));
      const float mnew = fmaxf(mrow[j], t);
      const float corr = __expf(mrow[j] - mnew);
      const float p0 = __expf(sc0[j] - mnew);
      const float p1 = __expf(sc1[j] - mnew);
      float ssum = p0 + p1;
      ssum += __shfl_xor(ssum, 1, 32);
      ssum += __shfl_xor(ssum, 2, 32);
      ssum += __shfl_xor(ssum, 4, 32);
      ssum += __shfl_xor(ssum, 8, 32);
      lrow[j] = lrow[j] * corr + ssum;
      mrow[j] = mnew;
#pragma unroll
      for (int c = 0; c < 4; c++) o[c][j] *= corr;
      // C-layout -> LDS row-major P[16][32]
      P[(j + hl * 8) * 32 + l15]      = (_Float16)p0;
      P[(j + hl * 8) * 32 + 16 + l15] = (_Float16)p1;
    }
    asm volatile("s_wait_dscnt 0" ::: "memory");
    __builtin_amdgcn_wave_barrier();
    // ---- reload P in A-fragment layout ----
    v16h pa;
    {
      const _Float16* pr = P + l15 * 32;
      v8h plo = *(const v8h*)(pr + hl * 16);
      v8h phi = *(const v8h*)(pr + 8 + hl * 16);
#pragma unroll
      for (int i = 0; i < 8; i++) { pa[i] = plo[i]; pa[8 + i] = phi[i]; }
    }
    // ---- o += P @ V  (V rows contiguous in keys) ----
#pragma unroll
    for (int c = 0; c < 4; c++) {
#ifdef USE_ASYNC
      const _Float16* vr = Vt + (c * 16 + l15) * 32 + hl * 16;
#else
      const _Float16* vr = V + (size_t)(c * 16 + l15) * SEQ + n0 + hl * 16;
#endif
      v16h vb = *(const v16h*)vr;
      o[c] = WMMA_F16(pa, vb, o[c]);
    }
  }

  // ---- normalize + store attn output row-major [4096][1024] ----
  const int b = bh >> 4;
#pragma unroll
  for (int c = 0; c < 4; c++) {
#pragma unroll
    for (int j = 0; j < 8; j++) {
      const int r = q0 + j + hl * 8;
      const int d = c * 16 + l15;
      oh[((size_t)(b * SEQ + r)) * DMODEL + h * DHEAD + d] =
          (_Float16)(o[c][j] / lrow[j]);
    }
  }
}

// ---------------------------------------------------------------------------
// Kernel 5: output projection [4096 x 1024] x [1024 x 1024] + bias -> f32
// ---------------------------------------------------------------------------
__global__ void oproj_kernel(const _Float16* __restrict__ ah,
                             const _Float16* __restrict__ wT,
                             const float* __restrict__ bias,
                             float* __restrict__ out) {
  const int NT = 1024 / 64;                       // 16 N-tiles
  const int w = blockIdx.x * 8 + (threadIdx.x >> 5);
  const int lane = threadIdx.x & 31;
  const int tm = w / NT, tn = w - tm * NT;
  const int hl = lane >> 4, l15 = lane & 15;
  const _Float16* Arow = ah + (size_t)(tm * 16 + l15) * DMODEL;
  v8f acc[4] = {};
  for (int k0 = 0; k0 < DMODEL; k0 += 32) {
    v16h a;
    v8h alo = *(const v8h*)(Arow + k0 + hl * 16);
    v8h ahi = *(const v8h*)(Arow + k0 + 8 + hl * 16);
#pragma unroll
    for (int i = 0; i < 8; i++) { a[i] = alo[i]; a[8 + i] = ahi[i]; }
#pragma unroll
    for (int c = 0; c < 4; c++) {
      const int col = tn * 64 + c * 16 + l15;
      v16h b = *(const v16h*)(wT + (size_t)col * DMODEL + k0 + hl * 16);
      acc[c] = WMMA_F16(a, b, acc[c]);
    }
  }
#pragma unroll
  for (int c = 0; c < 4; c++) {
    const int col = tn * 64 + c * 16 + l15;
    const float bv = bias[col];
#pragma unroll
    for (int j = 0; j < 8; j++) {
      const int r = tm * 16 + j + hl * 8;
      out[(size_t)r * DMODEL + col] = acc[c][j] + bv;
    }
  }
}

// ---------------------------------------------------------------------------
// Launch
// ---------------------------------------------------------------------------
extern "C" void kernel_launch(void* const* d_in, const int* in_sizes, int n_in,
                              void* d_out, int out_size, void* d_ws, size_t ws_size,
                              hipStream_t stream) {
  const float* x     = (const float*)d_in[0];
  const float* alibi = (const float*)d_in[1];
  const float* gamma = (const float*)d_in[2];
  const float* beta  = (const float*)d_in[3];
  const float* wqkv  = (const float*)d_in[4];
  const float* wout  = (const float*)d_in[5];
  const float* bout  = (const float*)d_in[6];
  float* out = (float*)d_out;

  char* ws = (char*)d_ws;
  _Float16* xn    = (_Float16*)ws;  ws += (size_t)ROWS * DMODEL * 2;        // 8 MB
  _Float16* wqkvT = (_Float16*)ws;  ws += (size_t)3 * DMODEL * DMODEL * 2;  // 6 MB
  _Float16* woutT = (_Float16*)ws;  ws += (size_t)DMODEL * DMODEL * 2;      // 2 MB
  _Float16* qh    = (_Float16*)ws;  ws += (size_t)ROWS * DMODEL * 2;        // 8 MB
  _Float16* kh    = (_Float16*)ws;  ws += (size_t)ROWS * DMODEL * 2;        // 8 MB
  _Float16* vt    = (_Float16*)ws;  ws += (size_t)ROWS * DMODEL * 2;        // 8 MB
  _Float16* ah    = (_Float16*)ws;  ws += (size_t)ROWS * DMODEL * 2;        // 8 MB

  ln_kernel<<<ROWS, 256, 0, stream>>>(x, gamma, beta, xn);
  convT_kernel<<<(3 * DMODEL * DMODEL + 255) / 256, 256, 0, stream>>>(
      wqkv, wqkvT, DMODEL, 3 * DMODEL);
  convT_kernel<<<(DMODEL * DMODEL + 255) / 256, 256, 0, stream>>>(
      wout, woutT, DMODEL, DMODEL);
  // 256 M-tiles * 48 N-tiles = 12288 wave-tiles / 8 waves per block
  qkv_kernel<<<1536, 256, 0, stream>>>(xn, wqkvT, qh, kh, vt);
  // 2*16*128 = 4096 (b,h,qtile) waves / 2 waves per block
  attn_kernel<<<2048, 64, 0, stream>>>(qh, kh, vt, alibi, ah);
  // 256 M-tiles * 16 N-tiles = 4096 wave-tiles / 8 waves per block
  oproj_kernel<<<512, 256, 0, stream>>>(ah, woutT, bout, out);
}
